// GRU_19885698581228
// MI455X (gfx1250) — matrix operand order repaired and checked by reading, not compiled, not run on hardware
//
#include <hip/hip_runtime.h>
#include <math.h>

// ---------------- types ----------------
typedef __attribute__((ext_vector_type(16))) __bf16 v16bf;
typedef __attribute__((ext_vector_type(8)))  float  v8f;

#define TT   1024   // time steps
#define BB   32     // batch
#define HH   512    // hidden
#define II   128    // input size
#define OO   128    // output size
#define GG   1536   // 3*H

// ---------------- helpers ----------------
static __device__ __forceinline__ unsigned short f2bf(float f) {
    union { float f; unsigned u; } v; v.f = f;
    unsigned r = v.u + 0x7FFFu + ((v.u >> 16) & 1u);   // round-nearest-even
    return (unsigned short)(r >> 16);
}
static __device__ __forceinline__ float bf2f(unsigned short h) {
    union { unsigned u; float f; } v; v.u = ((unsigned)h) << 16;
    return v.f;
}

union AFrag { v16bf v; unsigned u[8]; };
union BFrag { v16bf v; uint4 q[2]; };

// A fragment (16x32 bf16, MxK): lane 0-15: M=lane, K=0..15;
// lane 16-31: M=lane-16, K=16..31. Merges into two global_load_b128.
static __device__ __forceinline__ v16bf load_a_frag(const unsigned short* arow, int kbase, int lane) {
    AFrag a;
    int base = kbase + ((lane < 16) ? 0 : 16);
    a.u[0] = *(const unsigned*)(arow + base + 0);
    a.u[1] = *(const unsigned*)(arow + base + 2);
    a.u[2] = *(const unsigned*)(arow + base + 4);
    a.u[3] = *(const unsigned*)(arow + base + 6);
    a.u[4] = *(const unsigned*)(arow + base + 8);
    a.u[5] = *(const unsigned*)(arow + base + 10);
    a.u[6] = *(const unsigned*)(arow + base + 12);
    a.u[7] = *(const unsigned*)(arow + base + 14);
    return a.v;
}

// B fragment (32x16 bf16, KxN). B[k][n] = W[n0+n, k0+k] (W row-major NxK).
// lane 0-15: n=lane, K=0..15; lane 16-31: n=lane-16, K=16..31 (32 contiguous bytes).
static __device__ __forceinline__ v16bf load_b_frag(const unsigned short* wrow, int kbase, int lane) {
    BFrag b;
    int cb = kbase + ((lane < 16) ? 0 : 16);
    b.q[0] = *(const uint4*)(wrow + cb);
    b.q[1] = *(const uint4*)(wrow + cb + 8);
    return b.v;
}

// ---------------- conversion kernels ----------------
__global__ void cvt_f32_to_bf16(const float* __restrict__ in, unsigned short* __restrict__ out, int n) {
    for (int i = blockIdx.x * blockDim.x + threadIdx.x; i < n; i += gridDim.x * blockDim.x)
        out[i] = f2bf(in[i]);
}

// x: (B,T,I) f32  ->  x_bf: (T,B,I) bf16  (rows ordered r = t*B + b)
__global__ void cvt_x_transpose(const float* __restrict__ x, unsigned short* __restrict__ out) {
    int n = BB * TT * II;
    for (int i = blockIdx.x * blockDim.x + threadIdx.x; i < n; i += gridDim.x * blockDim.x) {
        int ii = i % II;
        int t  = (i / II) % TT;
        int b  = i / (II * TT);
        out[((size_t)t * BB + b) * II + ii] = f2bf(x[i]);
    }
}

// ---------------- bf16 WMMA GEMM:  C = A * W^T + bias ----------------
// A: (M,K) bf16 row-major.  W: (N,K) bf16 row-major.
// Each wave computes a 16x64 strip: one A fragment feeds 4 WMMAs (N blocking).
// mode 0: store bf16 row-major (ldc = N).  mode 1: f32 with (t*B+b)->(b,t) remap, N=OO.
__global__ void wmma_gemm_bf16(const unsigned short* __restrict__ A,
                               const unsigned short* __restrict__ W,
                               const float* __restrict__ bias,
                               unsigned short* __restrict__ Cbf,
                               float* __restrict__ Cf,
                               int M, int N, int K, int mode) {
    int lane   = threadIdx.x & 31;
    int waveId = blockIdx.x * (blockDim.x >> 5) + (threadIdx.x >> 5);
    int tilesM = M >> 4;
    int groupsN = N >> 6;                       // N tiles of 64
    if (waveId >= tilesM * groupsN) return;
    int tm  = waveId % tilesM;
    int tn4 = waveId / tilesM;

    const unsigned short* arow = A + (size_t)(tm * 16 + (lane & 15)) * K;
    const unsigned short* wrow0 = W + (size_t)(tn4 * 64 +  0 + (lane & 15)) * K;
    const unsigned short* wrow1 = W + (size_t)(tn4 * 64 + 16 + (lane & 15)) * K;
    const unsigned short* wrow2 = W + (size_t)(tn4 * 64 + 32 + (lane & 15)) * K;
    const unsigned short* wrow3 = W + (size_t)(tn4 * 64 + 48 + (lane & 15)) * K;

    v8f acc0 = {0.f,0.f,0.f,0.f,0.f,0.f,0.f,0.f};
    v8f acc1 = acc0, acc2 = acc0, acc3 = acc0;

    int ksteps = K >> 5;
    for (int k0 = 0; k0 < ksteps; ++k0) {
        int kb = k0 * 32;
        v16bf af = load_a_frag(arow, kb, lane);
        v16bf b0 = load_b_frag(wrow0, kb, lane);
        v16bf b1 = load_b_frag(wrow1, kb, lane);
        v16bf b2 = load_b_frag(wrow2, kb, lane);
        v16bf b3 = load_b_frag(wrow3, kb, lane);
        acc0 = __builtin_amdgcn_wmma_f32_16x16x32_bf16(false, af, false, b0, (short)0, acc0, false, false);
        acc1 = __builtin_amdgcn_wmma_f32_16x16x32_bf16(false, af, false, b1, (short)0, acc1, false, false);
        acc2 = __builtin_amdgcn_wmma_f32_16x16x32_bf16(false, af, false, b2, (short)0, acc2, false, false);
        acc3 = __builtin_amdgcn_wmma_f32_16x16x32_bf16(false, af, false, b3, (short)0, acc3, false, false);
    }

    int mbase = tm * 16 + ((lane < 16) ? 0 : 8);
#pragma unroll
    for (int sub = 0; sub < 4; ++sub) {
        v8f acc = (sub == 0) ? acc0 : (sub == 1) ? acc1 : (sub == 2) ? acc2 : acc3;
        int n = tn4 * 64 + sub * 16 + (lane & 15);
        float bn = bias[n];
#pragma unroll
        for (int v = 0; v < 8; ++v) {
            int m = mbase + v;
            float val = acc[v] + bn;
            if (mode == 0) {
                Cbf[(size_t)m * N + n] = f2bf(val);
            } else {
                int t = m >> 5, b = m & 31;          // m = t*BB + b
                Cf[((size_t)b * TT + t) * OO + n] = val;
            }
        }
    }
}

// ---------------- persistent weight-stationary GRU recurrence ----------------
// grid: 32 workgroups (16-column slice of H each), block: 256 threads = 8 waves.
// waves 0-5 do WMMA (gate = w/2 in {r,z,n}, mtile = w&1); all 8 waves do gate math.
// Each WMMA wave's Whh slice lives in registers: 16 B fragments = 128 VGPRs.
__global__ void __launch_bounds__(256, 1)
gru_recurrent(const unsigned short* __restrict__ gx,      // (T*B, 3H) bf16, includes bih
              const unsigned short* __restrict__ whh_bf,  // (3H, H) bf16
              const float* __restrict__ bhh,              // (3H)
              unsigned short* __restrict__ y_bf,          // (T*B, H) bf16 out
              float* __restrict__ hf0, float* __restrict__ hf1,
              unsigned short* __restrict__ hb0, unsigned short* __restrict__ hb1,
              unsigned* __restrict__ sync,
              float* __restrict__ hidden_out) {            // (B,H) final state
    __shared__ float gh[3][BB][16];

    const int tid   = threadIdx.x;
    const int lane  = tid & 31;
    const int wave  = tid >> 5;
    const int gate  = wave >> 1;                  // valid for waves 0-5
    const int mtile = wave & 1;
    const int j0 = blockIdx.x * 16;
    const bool mma_wave = (wave < 6);

    // ---- preload Whh B-fragments into registers (weight-stationary) ----
    v16bf bfrag[16];
    if (mma_wave) {
        const unsigned short* wrow = whh_bf + (size_t)(gate * HH + j0 + (lane & 15)) * HH;
#pragma unroll
        for (int k0 = 0; k0 < 16; ++k0)
            bfrag[k0] = load_b_frag(wrow, k0 * 32, lane);
    }

    // ---- loop-invariant gate-math state: fixed column j, rows b1 and b1+16 ----
    const int jj   = tid & 15;
    const int jcol = j0 + jj;
    const int b1   = tid >> 4;                    // 0..15
    const int b2   = b1 + 16;
    const float bh_r = bhh[jcol];
    const float bh_z = bhh[HH + jcol];
    const float bh_n = bhh[2 * HH + jcol];

    for (int t = 0; t < TT; ++t) {
        const unsigned short* hin_bf  = (t & 1) ? hb1 : hb0;
        const float*          hin_f   = (t & 1) ? hf1 : hf0;
        unsigned short*       hout_bf = (t & 1) ? hb0 : hb1;
        float*                hout_f  = (t & 1) ? hf0 : hf1;

        // ---- gh tile = h(16x512) * Whh_slice^T ----
        if (mma_wave) {
            const unsigned short* arow = hin_bf + (size_t)(mtile * 16 + (lane & 15)) * HH;
            v8f acc = {0.f,0.f,0.f,0.f,0.f,0.f,0.f,0.f};
#pragma unroll
            for (int k0 = 0; k0 < 16; ++k0) {
                v16bf af = load_a_frag(arow, k0 * 32, lane);
                acc = __builtin_amdgcn_wmma_f32_16x16x32_bf16(false, af, false, bfrag[k0],
                                                              (short)0, acc, false, false);
            }
            int n  = lane & 15;
            int mb = mtile * 16 + ((lane < 16) ? 0 : 8);
#pragma unroll
            for (int v = 0; v < 8; ++v)
                gh[gate][mb + v][n] = acc[v];
        }
        __syncthreads();

        // ---- prefetch next step's gx slice into cache ----
        if (t + 1 < TT && tid < 96) {
            int b = tid & 31, g = tid >> 5;
            __builtin_prefetch(gx + ((size_t)(t + 1) * BB + b) * GG + g * HH + j0, 0, 1);
        }

        // ---- elementwise gate math (f32): 2 rows per thread, same column ----
        size_t row1 = (size_t)t * BB + b1;
        size_t row2 = row1 + 16;
#pragma unroll
        for (int half = 0; half < 2; ++half) {
            int    b   = half ? b2 : b1;
            size_t row = half ? row2 : row1;
            float gxr = bf2f(gx[row * GG + jcol]);
            float gxz = bf2f(gx[row * GG + HH + jcol]);
            float gxn = bf2f(gx[row * GG + 2 * HH + jcol]);
            float ghr = gh[0][b][jj] + bh_r;
            float ghz = gh[1][b][jj] + bh_z;
            float ghn = gh[2][b][jj] + bh_n;
            float r  = 1.f / (1.f + expf(-(gxr + ghr)));
            float z  = 1.f / (1.f + expf(-(gxz + ghz)));
            float nn = tanhf(gxn + r * ghn);
            float hold = hin_f[b * HH + jcol];
            float hnew = (1.f - z) * nn + z * hold;
            hout_f[b * HH + jcol]  = hnew;
            hout_bf[b * HH + jcol] = f2bf(hnew);
            y_bf[row * HH + jcol]  = f2bf(hnew);
            if (t == TT - 1) hidden_out[b * HH + jcol] = hnew;
        }

        // ---- grid-wide barrier (cumulative counter, 32 workgroups) ----
        __threadfence();
        __syncthreads();
        if (tid == 0) {
            atomicAdd(sync, 1u);
            unsigned target = 32u * (unsigned)(t + 1);
            volatile unsigned* vs = sync;
            while (*vs < target) { __builtin_amdgcn_s_sleep(1); }
        }
        __syncthreads();
        __threadfence();
    }
}

// ---------------- host side ----------------
extern "C" void kernel_launch(void* const* d_in, const int* in_sizes, int n_in,
                              void* d_out, int out_size, void* d_ws, size_t ws_size,
                              hipStream_t stream) {
    const float* x    = (const float*)d_in[0];
    const float* Wih0 = (const float*)d_in[1];
    const float* Whh0 = (const float*)d_in[2];
    const float* bih0 = (const float*)d_in[3];
    const float* bhh0 = (const float*)d_in[4];
    const float* Wih1 = (const float*)d_in[5];
    const float* Whh1 = (const float*)d_in[6];
    const float* bih1 = (const float*)d_in[7];
    const float* bhh1 = (const float*)d_in[8];
    const float* Wout = (const float*)d_in[9];
    const float* bout = (const float*)d_in[10];

    char* ws = (char*)d_ws;
    size_t o = 0;
    auto alloc = [&](size_t bytes) { size_t r = o; o += (bytes + 255) & ~(size_t)255; return r; };
    size_t off_sync = alloc(256);
    size_t off_hf0  = alloc((size_t)BB * HH * 4);
    size_t off_hf1  = alloc((size_t)BB * HH * 4);
    size_t off_hb0  = alloc((size_t)BB * HH * 2);
    size_t off_hb1  = alloc((size_t)BB * HH * 2);
    size_t headBytes = o;                                  // zeroed before each layer
    size_t off_wih0 = alloc((size_t)GG * II * 2);
    size_t off_whh0 = alloc((size_t)GG * HH * 2);
    size_t off_wih1 = alloc((size_t)GG * HH * 2);
    size_t off_whh1 = alloc((size_t)GG * HH * 2);
    size_t off_wout = alloc((size_t)OO * HH * 2);
    size_t off_xbf  = alloc((size_t)TT * BB * II * 2);
    size_t off_gx   = alloc((size_t)TT * BB * GG * 2);
    size_t off_y0   = alloc((size_t)TT * BB * HH * 2);
    size_t off_y1   = alloc((size_t)TT * BB * HH * 2);

    unsigned*       syncp = (unsigned*)(ws + off_sync);
    float*          hf0   = (float*)(ws + off_hf0);
    float*          hf1   = (float*)(ws + off_hf1);
    unsigned short* hb0   = (unsigned short*)(ws + off_hb0);
    unsigned short* hb1   = (unsigned short*)(ws + off_hb1);
    unsigned short* wih0b = (unsigned short*)(ws + off_wih0);
    unsigned short* whh0b = (unsigned short*)(ws + off_whh0);
    unsigned short* wih1b = (unsigned short*)(ws + off_wih1);
    unsigned short* whh1b = (unsigned short*)(ws + off_whh1);
    unsigned short* woutb = (unsigned short*)(ws + off_wout);
    unsigned short* xbf   = (unsigned short*)(ws + off_xbf);
    unsigned short* gxb   = (unsigned short*)(ws + off_gx);
    unsigned short* y0b   = (unsigned short*)(ws + off_y0);
    unsigned short* y1b   = (unsigned short*)(ws + off_y1);

    float* outp = (float*)d_out;                       // (B,T,O)
    float* hid0 = outp + (size_t)BB * TT * OO;         // layer0 final h (B,H)
    float* hid1 = hid0 + (size_t)BB * HH;              // layer1 final h (B,H)

    const int M = TT * BB;                             // 32768 rows

    // 1) zero sync counter + h state
    hipMemsetAsync(ws, 0, headBytes, stream);

    // 2) convert inputs/weights to bf16
    cvt_x_transpose<<<4096, 256, 0, stream>>>(x, xbf);
    cvt_f32_to_bf16<<<768,  256, 0, stream>>>(Wih0, wih0b, GG * II);
    cvt_f32_to_bf16<<<3072, 256, 0, stream>>>(Whh0, whh0b, GG * HH);
    cvt_f32_to_bf16<<<3072, 256, 0, stream>>>(Wih1, wih1b, GG * HH);
    cvt_f32_to_bf16<<<3072, 256, 0, stream>>>(Whh1, whh1b, GG * HH);
    cvt_f32_to_bf16<<<256,  256, 0, stream>>>(Wout, woutb, OO * HH);

    // 3) gx0 = x * Wih0^T + bih0   (M x 1536, K=128)
    {
        int groups = (M / 16) * (GG / 64);
        wmma_gemm_bf16<<<(groups + 7) / 8, 256, 0, stream>>>(
            xbf, wih0b, bih0, gxb, nullptr, M, GG, II, 0);
    }

    // 4) layer-0 recurrence (persistent, weight-stationary)
    gru_recurrent<<<32, 256, 0, stream>>>(gxb, whh0b, bhh0, y0b,
                                          hf0, hf1, hb0, hb1, syncp, hid0);

    // 5) reset sync + h, gx1 = y0 * Wih1^T + bih1  (K=512)
    hipMemsetAsync(ws, 0, headBytes, stream);
    {
        int groups = (M / 16) * (GG / 64);
        wmma_gemm_bf16<<<(groups + 7) / 8, 256, 0, stream>>>(
            y0b, wih1b, bih1, gxb, nullptr, M, GG, HH, 0);
    }

    // 6) layer-1 recurrence
    gru_recurrent<<<32, 256, 0, stream>>>(gxb, whh1b, bhh1, y1b,
                                          hf0, hf1, hb0, hb1, syncp, hid1);

    // 7) out = y1 * Wout^T + bout, remapped (t,b) -> (b,t)
    {
        int groups = (M / 16) * (OO / 64);
        wmma_gemm_bf16<<<(groups + 7) / 8, 256, 0, stream>>>(
            y1b, woutb, bout, nullptr, outp, M, OO, HH, 1);
    }
    (void)in_sizes; (void)n_in; (void)out_size; (void)ws_size;
}
